// CL_Block_38457137168450
// MI455X (gfx1250) — matrix operand order repaired and checked by reading, not compiled, not run on hardware
//
#include <hip/hip_runtime.h>
#include <math.h>

#define DEV __device__ __forceinline__

typedef unsigned short bf16u;
typedef __attribute__((ext_vector_type(16))) __bf16 v16bf;
typedef __attribute__((ext_vector_type(8)))  float  v8f;

constexpr int NB  = 8;     // batch
constexpr int NT  = 2000;  // tokens (points)
constexpr int NTP = 2016;  // tokens padded to 63*32 for K-loops over tokens
constexpr int NC  = 128;   // channels
constexpr int NH  = 4;     // heads
constexpr int JT  = 5;     // token tiles per wave in k_gemm (16*5 = 80; 2000 = 25*80)

// ---------- small helpers ----------
DEV bf16u f2bf(float f){ unsigned u=__float_as_uint(f); return (bf16u)((u + 0x7FFFu + ((u>>16)&1u))>>16); }
DEV float bf2f(bf16u v){ return __uint_as_float(((unsigned)v)<<16); }

union FragU { uint4 q[2]; v16bf v; };
DEV v16bf ldfrag(const bf16u* lo, const bf16u* hi){
  FragU f; f.q[0]=*(const uint4*)lo; f.q[1]=*(const uint4*)hi; return f.v;
}
DEV v8f vzero(){ v8f z;
#pragma unroll
  for(int i=0;i<8;i++) z[i]=0.f;
  return z; }
DEV v8f wmma_bf(v16bf a, v16bf b, v8f c){
  // D(16x16,f32) = A(16x32,bf16) x B(32x16,bf16) + C
  return __builtin_amdgcn_wmma_f32_16x16x32_bf16(false,a,false,b,(short)0,c,false,false);
}
DEV uint4 pack8(const float* t){
  uint4 o;
  o.x=(unsigned)f2bf(t[0]) | ((unsigned)f2bf(t[1])<<16);
  o.y=(unsigned)f2bf(t[2]) | ((unsigned)f2bf(t[3])<<16);
  o.z=(unsigned)f2bf(t[4]) | ((unsigned)f2bf(t[5])<<16);
  o.w=(unsigned)f2bf(t[6]) | ((unsigned)f2bf(t[7])<<16);
  return o;
}

// ---------- f32 -> bf16 weight conversion ----------
__global__ void k_cvt(const float* __restrict__ s, bf16u* __restrict__ d, int n){
  int i = blockIdx.x*256 + threadIdx.x;
  if (i < n) d[i] = f2bf(s[i]);
}

// ---------- embed conv 4->128, token-major f32 out ----------
__global__ void k_embed(const float* __restrict__ x, const float* __restrict__ w,
                        const float* __restrict__ bias, float* __restrict__ out){
  int b = blockIdx.x, c = threadIdx.x;            // 128 threads
  float w0=w[c*4+0], w1=w[c*4+1], w2=w[c*4+2], w3=w[c*4+3], bb=bias[c];
  const float* xb = x + (size_t)b*4*NT;
  for (int n=0;n<NT;n++){
    float v = bb + w0*xb[n] + w1*xb[NT+n] + w2*xb[2*NT+n] + w3*xb[3*NT+n];
    out[((size_t)b*NT+n)*NC + c] = v;
  }
}

// ---------- InstanceNorm(eps) -> BN(g,bb) [-> +res] [-> relu], writes f32 + bf16 ----------
__global__ void k_norm(const float* __restrict__ src, float* __restrict__ dstf,
                       bf16u* __restrict__ dstb, const float* __restrict__ res,
                       const float* __restrict__ gam, const float* __restrict__ beta,
                       float eps, int relu){
  int b = blockIdx.x, c = threadIdx.x;            // 128 threads
  const float* p = src + (size_t)b*NT*NC + c;
  float s=0.f, s2=0.f;
  for (int n=0;n<NT;n++){ float v=p[(size_t)n*NC]; s+=v; s2+=v*v; }
  float m = s/NT, var = s2/NT - m*m;
  float rstd = rsqrtf(fmaxf(var,0.f)+eps);
  float sc = gam[c]*rsqrtf(1.f+1e-5f)*rstd;       // BN eval: g/sqrt(1+eps_bn)
  float sh = beta[c] - m*sc;
  for (int n=0;n<NT;n++){
    float v = p[(size_t)n*NC]*sc + sh;
    if (res) v += res[((size_t)b*NT+n)*NC + c];
    if (relu) v = fmaxf(v,0.f);
    size_t o = ((size_t)b*NT+n)*NC + c;
    dstf[o] = v; dstb[o] = f2bf(v);
  }
}

// ---------- generic conv GEMM: out[b][n][m] = epi(alpha*(W@act) + bias) ----------
// one wave per (16 couts x 80 tokens): weight A-fragment reused across 5 WMMAs
__global__ void k_gemm(const bf16u* __restrict__ W, const bf16u* __restrict__ act,
                       const float* __restrict__ bias, const float* __restrict__ gam,
                       const float* __restrict__ beta, const float* __restrict__ res,
                       float* __restrict__ outf, bf16u* __restrict__ outb,
                       int M, int K, float alpha, int relu){
  const int lane=threadIdx.x, lm=lane&15, lh=lane>>4;
  const int n0=blockIdx.x*16*JT, m0=blockIdx.y*16, b=blockIdx.z;
  const bf16u* wrow = W + (size_t)(m0 + lm)*K;
  const bf16u* arow[JT];
  v8f acc[JT];
#pragma unroll
  for (int j=0;j<JT;j++){
    arow[j] = act + ((size_t)b*NT + n0 + j*16 + lm)*K;
    acc[j]  = vzero();
  }
  for (int f=0; f<K; f+=32){
    v16bf A = ldfrag(wrow + f + lh*8, wrow + f + 16 + lh*8);
#pragma unroll
    for (int j=0;j<JT;j++){
      v16bf Bv = ldfrag(arow[j] + f + lh*8, arow[j] + f + 16 + lh*8);
      acc[j] = wmma_bf(A, Bv, acc[j]);
    }
  }
  const int mb = m0 + 8*lh;
  const float rbn = rsqrtf(1.f+1e-5f);
  float bi[8]; *(float4*)&bi[0]=*(const float4*)(bias+mb); *(float4*)&bi[4]=*(const float4*)(bias+mb+4);
  float g[8], be[8];
  if (gam){
    *(float4*)&g[0]=*(const float4*)(gam+mb);  *(float4*)&g[4]=*(const float4*)(gam+mb+4);
    *(float4*)&be[0]=*(const float4*)(beta+mb);*(float4*)&be[4]=*(const float4*)(beta+mb+4);
  }
#pragma unroll
  for (int j=0;j<JT;j++){
    const int tok = n0 + j*16 + lm;
    float t[8];
#pragma unroll
    for (int r=0;r<8;r++) t[r] = alpha*acc[j][r] + bi[r];
    if (gam){
#pragma unroll
      for (int r=0;r<8;r++) t[r] = t[r]*(g[r]*rbn) + be[r];
    }
    if (res){
      float rr[8];
      const float* rp = res + ((size_t)b*NT+tok)*M + mb;
      *(float4*)&rr[0]=*(const float4*)rp; *(float4*)&rr[4]=*(const float4*)(rp+4);
#pragma unroll
      for (int r=0;r<8;r++) t[r] += rr[r];
    }
    if (relu){
#pragma unroll
      for (int r=0;r<8;r++) t[r] = fmaxf(t[r],0.f);
    }
    float* op = outf + ((size_t)b*NT+tok)*M + mb;
    *(float4*)op = *(const float4*)&t[0]; *(float4*)(op+4) = *(const float4*)&t[4];
    if (outb) *(uint4*)(outb + ((size_t)b*NT+tok)*M + mb) = pack8(t);
  }
}

// ---------- per-token sum of squares (for kNN keys) ----------
__global__ void k_rowsq(const bf16u* __restrict__ a, float* __restrict__ xx){
  int t = blockIdx.x*256 + threadIdx.x;
  if (t >= NB*NT) return;
  const bf16u* r = a + (size_t)t*NC; float s=0.f;
  for (int c=0;c<NC;c++){ float v=bf2f(r[c]); s+=v*v; }
  xx[t]=s;
}

// ---------- fused pairwise-distance GEMM + top-8 (wave per 16 query rows) ----------
__global__ void k_knn(const bf16u* __restrict__ a, const float* __restrict__ xx,
                      int* __restrict__ knn){
  __shared__ float tile[256];
  const int lane=threadIdx.x, lm=lane&15, lh=lane>>4;
  const int b=blockIdx.y, i0=blockIdx.x*16;
  const bf16u* arow = a + ((size_t)b*NT + i0 + lm)*NC;
  v16bf af[4];
#pragma unroll
  for (int f=0; f<4; ++f)
    af[f] = ldfrag(arow + f*32 + lh*8, arow + f*32 + 16 + lh*8);
  float vals[8]; int ids[8]; float vmin=-3e38f; int minp=0;
#pragma unroll
  for (int t=0;t<8;t++){ vals[t]=-3e38f; ids[t]=0; }
  for (int jt=0; jt<NT/16; ++jt){
    const int j0 = jt*16;
    const bf16u* brow = a + ((size_t)b*NT + j0 + lm)*NC;
    v8f acc = vzero();
#pragma unroll
    for (int f=0; f<4; ++f)
      acc = wmma_bf(af[f], ldfrag(brow + f*32 + lh*8, brow + f*32 + 16 + lh*8), acc);
    const float xj = xx[b*NT + j0 + lm];     // row-constant xx_i dropped: rank-invariant
#pragma unroll
    for (int r=0;r<8;r++) tile[(r+8*lh)*16 + lm] = 2.f*acc[r] - xj;
    __syncthreads();
    if (lane < 16){
      for (int c2=0;c2<16;c2++){
        float key = tile[lane*16+c2];
        if (key > vmin){
          vals[minp]=key; ids[minp]=j0+c2;
          vmin=vals[0]; minp=0;
#pragma unroll
          for (int t=1;t<8;t++) if (vals[t]<vmin){ vmin=vals[t]; minp=t; }
        }
      }
    }
    __syncthreads();
  }
  if (lane < 16){
    int* o = knn + ((size_t)b*NT + i0 + lane)*8;
#pragma unroll
    for (int t=0;t<8;t++) o[t]=ids[t];
  }
}

// ---------- fused edge features + conv(256->128)+BN+ReLU + conv(128->128)+BN+ReLU + max_k ----------
__global__ __launch_bounds__(256,1)
void k_dgcnn(const bf16u* __restrict__ src, const int* __restrict__ knn,
             const bf16u* __restrict__ w1, const float* __restrict__ b1,
             const float* __restrict__ g1, const float* __restrict__ bb1,
             const bf16u* __restrict__ w2, const float* __restrict__ b2,
             const float* __restrict__ g2, const float* __restrict__ bb2,
             float* __restrict__ outf, bf16u* __restrict__ outb){
  __shared__ bf16u smE[8*16*256];   // per-wave 16 edge rows x 256 ch
  __shared__ bf16u smH[8*16*128];   // per-wave conv1 result
  const int tid=threadIdx.x, w=tid>>5, lane=tid&31, lm=lane&15, lh=lane>>4;
  const int b=blockIdx.y, pbase=blockIdx.x*16 + 2*w;   // wave handles 2 points
  bf16u* E  = smE + w*(16*256);
  bf16u* Hh = smH + w*(16*128);
  for (int row=0; row<16; ++row){
    int pp=row>>3, kk=row&7, tc=pbase+pp;
    int tn = knn[((size_t)b*NT+tc)*8 + kk];
    const bf16u* cr = src + ((size_t)b*NT+tc)*NC;
    const bf16u* nr = src + ((size_t)b*NT+tn)*NC;
    int c0 = lane*4;
#pragma unroll
    for (int i2=0;i2<4;i2++){
      bf16u cv=cr[c0+i2], nv=nr[c0+i2];
      E[row*256 + c0+i2]       = cv;
      E[row*256 + 128 + c0+i2] = f2bf(bf2f(cv)-bf2f(nv));
    }
  }
  __syncthreads();
  const float rbn = rsqrtf(1.f+1e-5f);
  v16bf af[8];
#pragma unroll
  for (int f=0; f<8; ++f)
    af[f] = ldfrag(E + lm*256 + f*32 + lh*8, E + lm*256 + f*32 + 16 + lh*8);
  for (int ct=0; ct<8; ++ct){
    v8f acc = vzero();
    const bf16u* wr = w1 + (size_t)(ct*16+lm)*256;
#pragma unroll
    for (int f=0; f<8; ++f)
      acc = wmma_bf(af[f], ldfrag(wr + f*32 + lh*8, wr + f*32 + 16 + lh*8), acc);
    int c = ct*16+lm;
    float bb=b1[c], sc=g1[c]*rbn, sh=bb1[c];
#pragma unroll
    for (int r=0;r<8;r++)
      Hh[(r+8*lh)*128 + c] = f2bf(fmaxf((acc[r]+bb)*sc+sh, 0.f));
  }
  __syncthreads();
  v16bf a2[4];
#pragma unroll
  for (int f=0; f<4; ++f)
    a2[f] = ldfrag(Hh + lm*128 + f*32 + lh*8, Hh + lm*128 + f*32 + 16 + lh*8);
  for (int ct=0; ct<8; ++ct){
    v8f acc = vzero();
    const bf16u* wr = w2 + (size_t)(ct*16+lm)*128;
#pragma unroll
    for (int f=0; f<4; ++f)
      acc = wmma_bf(a2[f], ldfrag(wr + f*32 + lh*8, wr + f*32 + 16 + lh*8), acc);
    int c = ct*16+lm;
    float bb=b2[c], sc=g2[c]*rbn, sh=bb2[c];
    float mx=-3e38f;
#pragma unroll
    for (int r=0;r<8;r++) mx = fmaxf(mx, fmaxf((acc[r]+bb)*sc+sh, 0.f));  // max over 8 neighbors
    int tok = pbase + lh;
    size_t o = ((size_t)b*NT+tok)*NC + c;
    outf[o]=mx; outb[o]=f2bf(mx);
  }
}

// ---------- per-channel L2 normalize over tokens, write channel-major bf16 padded ----------
__global__ void k_l2norm(const float* __restrict__ src, bf16u* __restrict__ dst){
  int b=blockIdx.x, c=threadIdx.x;
  const float* p = src + (size_t)b*NT*NC + c;
  float s=0.f;
  for (int n=0;n<NT;n++){ float v=p[(size_t)n*NC]; s+=v*v; }
  float rn = 1.f/fmaxf(sqrtf(s),1e-12f);
  bf16u* d = dst + ((size_t)b*NC + c)*NTP;
  for (int n=0;n<NT;n++)  d[n] = f2bf(p[(size_t)n*NC]*rn);
  for (int n=NT;n<NTP;n++) d[n] = 0;
}

// ---------- channel attention: qk^T (K=2016 tokens), softmax, attn@v ----------
__global__ void k_attn(const bf16u* __restrict__ qcm, const bf16u* __restrict__ kcm,
                       const bf16u* __restrict__ vtb, const float* __restrict__ temp,
                       bf16u* __restrict__ outb){
  __shared__ float sAm[32*32];
  __shared__ bf16u sBm[32*32];
  const int lane=threadIdx.x, lm=lane&15, lh=lane>>4;
  const int hh=blockIdx.x, b=blockIdx.y;
  const bf16u* qb = qcm + ((size_t)b*NC + hh*32)*NTP;
  const bf16u* kb = kcm + ((size_t)b*NC + hh*32)*NTP;
  v8f a00=vzero(), a01=vzero(), a10=vzero(), a11=vzero();
  for (int kt=0; kt<NTP/32; ++kt){
    const int k0 = kt*32;
    v16bf A0 = ldfrag(qb + (size_t)lm*NTP      + k0 + lh*8,  qb + (size_t)lm*NTP      + k0 + 16 + lh*8);
    v16bf A1 = ldfrag(qb + (size_t)(16+lm)*NTP + k0 + lh*8,  qb + (size_t)(16+lm)*NTP + k0 + 16 + lh*8);
    v16bf B0 = ldfrag(kb + (size_t)lm*NTP      + k0 + lh*8,  kb + (size_t)lm*NTP      + k0 + 16 + lh*8);
    v16bf B1 = ldfrag(kb + (size_t)(16+lm)*NTP + k0 + lh*8,  kb + (size_t)(16+lm)*NTP + k0 + 16 + lh*8);
    a00=wmma_bf(A0,B0,a00); a01=wmma_bf(A0,B1,a01);
    a10=wmma_bf(A1,B0,a10); a11=wmma_bf(A1,B1,a11);
  }
  const float ts = temp[hh];
#pragma unroll
  for (int r=0;r<8;r++){
    sAm[(r+8*lh)*32 + lm]         = a00[r]*ts;
    sAm[(r+8*lh)*32 + 16+lm]      = a01[r]*ts;
    sAm[(16+r+8*lh)*32 + lm]      = a10[r]*ts;
    sAm[(16+r+8*lh)*32 + 16+lm]   = a11[r]*ts;
  }
  __syncthreads();
  { // softmax: one lane per row of the 32x32 attention map
    int rw = lane;
    float mx=-3e38f;
    for (int j=0;j<32;j++) mx = fmaxf(mx, sAm[rw*32+j]);
    float sm=0.f;
    for (int j=0;j<32;j++) sm += __expf(sAm[rw*32+j]-mx);
    float inv = 1.f/sm;
    for (int j=0;j<32;j++) sBm[rw*32+j] = f2bf(__expf(sAm[rw*32+j]-mx)*inv);
  }
  __syncthreads();
  v16bf at0 = ldfrag(sBm + lm*32      + lh*8, sBm + lm*32      + 16 + lh*8);
  v16bf at1 = ldfrag(sBm + (16+lm)*32 + lh*8, sBm + (16+lm)*32 + 16 + lh*8);
  for (int nt=0; nt<NT/16; ++nt){
    const int tok = nt*16 + lm;
    const bf16u* vr = vtb + ((size_t)b*NT+tok)*NC + hh*32;
    v16bf Bv = ldfrag(vr + lh*8, vr + 16 + lh*8);
    v8f o0 = wmma_bf(at0, Bv, vzero());
    v8f o1 = wmma_bf(at1, Bv, vzero());
    float t0[8], t1[8];
#pragma unroll
    for (int r=0;r<8;r++){ t0[r]=o0[r]; t1[r]=o1[r]; }
    bf16u* op = outb + ((size_t)b*NT+tok)*NC + hh*32;
    *(uint4*)(op + 8*lh)      = pack8(t0);
    *(uint4*)(op + 16 + 8*lh) = pack8(t1);
  }
}

// ---------- final 128->2 conv ----------
__global__ void k_outconv(const float* __restrict__ hg, const float* __restrict__ w,
                          const float* __restrict__ bias, float* __restrict__ out){
  int t = blockIdx.x*256 + threadIdx.x;
  if (t >= NB*NT) return;
  int b = t/NT, n = t%NT;
  const float* r = hg + (size_t)t*NC;
  float a0 = bias[0], a1 = bias[1];
  for (int c=0;c<NC;c++){ float v=r[c]; a0 += w[c]*v; a1 += w[NC+c]*v; }
  out[((size_t)b*2+0)*NT + n] = a0;
  out[((size_t)b*2+1)*NT + n] = a1;
}

// =====================================================================
extern "C" void kernel_launch(void* const* d_in, const int* in_sizes, int n_in,
                              void* d_out, int out_size, void* d_ws, size_t ws_size,
                              hipStream_t stream){
  (void)in_sizes; (void)n_in; (void)out_size; (void)ws_size;
  // ---- input pointer map (setup_inputs dict order) ----
  const float* x    = (const float*)d_in[0];
  const float* c0w  = (const float*)d_in[1];
  const float* c0b  = (const float*)d_in[2];
  const float* c0g  = (const float*)d_in[3];
  const float* c0bb = (const float*)d_in[4];
  const float* temp = (const float*)d_in[41];
  const float* projb= (const float*)d_in[43];
  const float* f1b_ = (const float*)d_in[45];
  const float* f1g  = (const float*)d_in[46];
  const float* f1bb = (const float*)d_in[47];
  const float* f2b_ = (const float*)d_in[49];
  const float* f2g  = (const float*)d_in[50];
  const float* f2bb = (const float*)d_in[51];
  const float* ggb  = (const float*)d_in[55];
  const float* ggg  = (const float*)d_in[56];
  const float* ggbb = (const float*)d_in[57];
  const float* outw = (const float*)d_in[58];
  const float* outbv= (const float*)d_in[59];

  // ---- workspace layout ----
  char* wsb = (char*)d_ws;
  size_t off = 0;
  auto alloc = [&](size_t bytes)->void*{ void* p = wsb + off; off = (off + bytes + 255) & ~(size_t)255; return p; };
  const size_t ACT  = (size_t)NB*NT*NC;
  bf16u* wbf   = (bf16u*)alloc(262144*sizeof(bf16u));   // all bf16 weights
  float* tmp0  = (float*)alloc(ACT*4);
  float* hF    = (float*)alloc(ACT*4);
  bf16u* hB    = (bf16u*)alloc(ACT*2);
  float* sAF   = (float*)alloc(ACT*4);
  bf16u* sAB   = (bf16u*)alloc(ACT*2);
  float* sMF   = (float*)alloc(ACT*4);
  bf16u* sMB   = (bf16u*)alloc(ACT*2);
  float* xx    = (float*)alloc((size_t)NB*NT*4);
  int*   idxb  = (int*)  alloc((size_t)NB*NT*8*4);
  bf16u* qcm   = (bf16u*)alloc((size_t)NB*NC*NTP*2);
  bf16u* kcm   = (bf16u*)alloc((size_t)NB*NC*NTP*2);
  bf16u* attno = (bf16u*)alloc(ACT*2);
  float* h2F   = (float*)alloc(ACT*4);
  bf16u* h2B   = (bf16u*)alloc(ACT*2);

  // ---- convert weights to bf16 ----
  const int wIdx[13] = {5,9,13, 17,21,25, 29,33,37, 42,44,48,54};
  const int wCnt[13] = {16384,32768,16384, 16384,32768,16384, 16384,32768,16384,
                        16384,16384,16384,16384};
  size_t woff[13]; { size_t c=0; for (int i=0;i<13;i++){ woff[i]=c; c+=wCnt[i]; } }
  for (int i=0;i<13;i++)
    k_cvt<<<(wCnt[i]+255)/256,256,0,stream>>>((const float*)d_in[wIdx[i]], wbf+woff[i], wCnt[i]);
  bf16u* wproj = wbf + woff[9];
  bf16u* wffn1 = wbf + woff[10];
  bf16u* wffn2 = wbf + woff[11];
  bf16u* wgg   = wbf + woff[12];

  const dim3 gG(NT/(16*JT), NC/16, NB);   // conv GEMM grid: 25 x 8 x 8
  // ---- embed: conv(4->128) -> inorm(1e-3)+BN+ReLU ----
  k_embed<<<NB,NC,0,stream>>>(x, c0w, c0b, tmp0);
  k_norm<<<NB,NC,0,stream>>>(tmp0, hF, hB, nullptr, c0g, c0bb, 1e-3f, 1);

  // ---- q/k/v streams ----
  for (int s=0; s<3; ++s){
    const int base = 5 + 12*s;
    const bf16u* wf  = wbf + woff[3*s+0];
    const bf16u* wg1 = wbf + woff[3*s+1];
    const bf16u* wg2 = wbf + woff[3*s+2];
    k_gemm<<<gG,32,0,stream>>>(wf, hB, (const float*)d_in[base+1], nullptr,nullptr,nullptr,
                               tmp0, nullptr, NC, NC, 1.f, 0);
    k_norm<<<NB,NC,0,stream>>>(tmp0, sAF, sAB, nullptr,
                               (const float*)d_in[base+2], (const float*)d_in[base+3], 1e-3f, 1);
    k_rowsq<<<(NB*NT+255)/256,256,0,stream>>>(sAB, xx);
    k_knn<<<dim3(NT/16,NB),32,0,stream>>>(sAB, xx, idxb);
    k_dgcnn<<<dim3(NT/16,NB),256,0,stream>>>(sAB, idxb,
        wg1, (const float*)d_in[base+5], (const float*)d_in[base+6], (const float*)d_in[base+7],
        wg2, (const float*)d_in[base+9], (const float*)d_in[base+10],(const float*)d_in[base+11],
        sMF, sMB);
    if (s==0)      k_l2norm<<<NB,NC,0,stream>>>(sMF, qcm);
    else if (s==1) k_l2norm<<<NB,NC,0,stream>>>(sMF, kcm);
    // s==2: sMB (bf16 token-major v) survives for attention
  }

  // ---- channel attention + proj + residual ----
  k_attn<<<dim3(NH,NB),32,0,stream>>>(qcm, kcm, sMB, temp, attno);
  k_gemm<<<gG,32,0,stream>>>(wproj, attno, projb, nullptr,nullptr, hF, h2F, h2B, NC, NC, 1.f, 0);

  // ---- FFN: conv -> inorm+BN+ReLU -> conv -> inorm+BN (+res, relu) ----
  k_gemm<<<gG,32,0,stream>>>(wffn1, h2B, f1b_, nullptr,nullptr,nullptr, tmp0, nullptr, NC, NC, 1.f, 0);
  k_norm<<<NB,NC,0,stream>>>(tmp0, sAF, sAB, nullptr, f1g, f1bb, 1e-5f, 1);
  k_gemm<<<gG,32,0,stream>>>(wffn2, sAB, f2b_, nullptr,nullptr,nullptr, tmp0, nullptr, NC, NC, 1.f, 0);
  k_norm<<<NB,NC,0,stream>>>(tmp0, sMF, hB, h2F, f2g, f2bb, 1e-5f, 1);   // h3 in (sMF,hB)

  // ---- gg_block: softmax rows sum to 1 exactly -> D == 2.0; wc logits are dead ----
  k_gemm<<<gG,32,0,stream>>>(wgg, hB, ggb, ggg, ggbb, nullptr, tmp0, nullptr, NC, NC, 2.f, 1);

  // ---- final 128->2 conv ----
  k_outconv<<<(NB*NT+255)/256,256,0,stream>>>(tmp0, outw, outbv, (float*)d_out);
}
